// CrossStageAttention_90340342104139
// MI455X (gfx1250) — matrix operand order repaired and checked by reading, not compiled, not run on hardware
//
#include <hip/hip_runtime.h>
#include <cstddef>

// ---------------- CDNA5 types / helpers ----------------
typedef __bf16 bf16;
typedef __attribute__((ext_vector_type(16))) __bf16 v16bf;
typedef __attribute__((ext_vector_type(8)))  __bf16 v8bf;
typedef __attribute__((ext_vector_type(8)))  float  v8f;

__device__ __forceinline__ bf16 f2bf(float f) {
  unsigned u = __builtin_bit_cast(unsigned, f);
  unsigned r = u + 0x7FFFu + ((u >> 16) & 1u);
  unsigned short h = (unsigned short)(r >> 16);
  return __builtin_bit_cast(bf16, h);
}
__device__ __forceinline__ float bf2f(bf16 b) {
  unsigned short h = __builtin_bit_cast(unsigned short, b);
  unsigned u = ((unsigned)h) << 16;
  return __builtin_bit_cast(float, u);
}
__device__ __forceinline__ v16bf zero16() {
  v16bf z;
#pragma unroll
  for (int i = 0; i < 16; ++i) z[i] = f2bf(0.0f);
  return z;
}

// A-fragment: 16(M) x 32(K) bf16, row-major source `base` points at (row0,k0); ld in elements.
// lanes 0-15: rows 0-15 with K[0..7],[16..23]; lanes 16-31: rows 0-15 with K[8..15],[24..31]
__device__ __forceinline__ v16bf load_a(const bf16* base, int ld, int lane) {
  int row = lane & 15;
  int kb  = (lane >> 4) << 3;
  const bf16* p = base + (size_t)row * ld + kb;
  v8bf lo = *(const v8bf*)(p);
  v8bf hi = *(const v8bf*)(p + 16);
  return __builtin_shufflevector(lo, hi, 0,1,2,3,4,5,6,7,8,9,10,11,12,13,14,15);
}
// B-fragment: stored [N][K] row-major; lane -> col = lane&15, K[kb..kb+7],[kb+16..kb+23], kb=(lane>>4)*8
__device__ __forceinline__ v16bf load_b(const bf16* base, int ld, int lane) {
  int col = lane & 15;
  int ko  = (lane >> 4) << 3;
  const bf16* p = base + (size_t)col * ld + ko;
  v8bf lo = *(const v8bf*)(p);
  v8bf hi = *(const v8bf*)(p + 16);
  return __builtin_shufflevector(lo, hi, 0,1,2,3,4,5,6,7,8,9,10,11,12,13,14,15);
}
__device__ __forceinline__ v8f wmma_bf(v16bf a, v16bf b, v8f c) {
  return __builtin_amdgcn_wmma_f32_16x16x32_bf16(false, a, false, b, (short)0, c, false, false);
}

#define C_DIM 512
#define N_TOK 1024
#define M_TOT 16384   // B*H*W = 16*1024
#define SCALE_ATTN 0.17677669529663687f  // 32^-0.5

// ---------------- weight conversion (fp32 -> bf16, with transposes) ----------------
__global__ void k_cvt(const float* wq, const float* wp, const float* fw, const float* ow,
                      bf16* wq_bf, bf16* wp_bf, bf16* fw_bf, bf16* ow_bf) {
  int stride = gridDim.x * blockDim.x;
  int t0 = blockIdx.x * blockDim.x + threadIdx.x;
  for (int i = t0; i < 512 * 512; i += stride) wq_bf[i] = f2bf(wq[i]);           // [c][k]
  for (int i = t0; i < 512 * 256; i += stride) wp_bf[i] = f2bf(wp[i]);           // [c][k]
  for (int i = t0; i < 512 * 1024; i += stride) {                                 // [o][i] <- [i][o]
    int o = i >> 10, k = i & 1023;
    fw_bf[i] = f2bf(fw[(size_t)k * 512 + o]);
  }
  for (int i = t0; i < 9 * 512 * 512; i += stride) {                              // [tap][o][i]
    int tap = i >> 18;
    int o = (i >> 9) & 511;
    int k = i & 511;
    ow_bf[i] = f2bf(ow[((size_t)tap * 512 + k) * 512 + o]);
  }
}

__global__ void k_zero(float* p, int n) {
  int i = blockIdx.x * blockDim.x + threadIdx.x;
  if (i < n) p[i] = 0.f;
}

// ---------------- qkv = x @ Wqkv^T  -> bf16 [16384][512] + transposed copy ----------------
__global__ __launch_bounds__(256) void k_gemm_qkv(const float* __restrict__ x,
                                                  const bf16* __restrict__ wq_bf,
                                                  bf16* __restrict__ qkv,
                                                  bf16* __restrict__ qkvT) {
  __shared__ bf16 As[16 * 512];
  int mt = blockIdx.x;  // 1024 M-tiles of 16
  int tid = threadIdx.x;
  for (int i = tid; i < 16 * 512; i += 256) {
    int row = i >> 9, k = i & 511;
    As[i] = f2bf(x[((size_t)mt * 16 + row) * 512 + k]);
  }
  __syncthreads();
  int wave = tid >> 5, lane = tid & 31;
  for (int t = 0; t < 4; ++t) {
    int nc = wave * 64 + t * 16;
    v8f acc = {0.f, 0.f, 0.f, 0.f, 0.f, 0.f, 0.f, 0.f};
    for (int ks = 0; ks < 16; ++ks) {
      v16bf a = load_a(As + ks * 32, 512, lane);
      v16bf b = load_b(wq_bf + (size_t)nc * 512 + ks * 32, 512, lane);
      acc = wmma_bf(a, b, acc);
    }
    int cc = nc + (lane & 15);
    int mb = mt * 16 + ((lane >> 4) << 3);
#pragma unroll
    for (int r = 0; r < 8; ++r) {
      int m = mb + r;
      bf16 v = f2bf(acc[r]);
      qkv[(size_t)m * 512 + cc] = v;
      qkvT[((size_t)(m >> 10) * 512 + cc) * 1024 + (m & 1023)] = v;
    }
  }
}

// ---------------- prevqkv GEMM fused with 2x2 avg/max pooling ----------------
__global__ __launch_bounds__(256) void k_gemm_pool(const float* __restrict__ prevx,
                                                   const bf16* __restrict__ wp_bf,
                                                   bf16* __restrict__ avg, bf16* __restrict__ avgT,
                                                   bf16* __restrict__ mx,  bf16* __restrict__ mxT) {
  __shared__ bf16 As[4][16 * 256];
  int blk = blockIdx.x;  // b*64 + pooled-row-tile
  int b = blk >> 6, pt = blk & 63;
  int tid = threadIdx.x;
  for (int idx = tid; idx < 4 * 16 * 256; idx += 256) {
    int q = idx >> 12, rem = idx & 4095, p = rem >> 8, k = rem & 255;
    int pg = pt * 16 + p;
    int pi = pg >> 5, pj = pg & 31;
    int dy = q >> 1, dx = q & 1;
    As[q][p * 256 + k] =
        f2bf(prevx[(((size_t)b * 64 + 2 * pi + dy) * 64 + (2 * pj + dx)) * 256 + k]);
  }
  __syncthreads();
  int wave = tid >> 5, lane = tid & 31;
  for (int t = 0; t < 4; ++t) {
    int nc = wave * 64 + t * 16;
    v8f c00 = {0.f,0.f,0.f,0.f,0.f,0.f,0.f,0.f};
    v8f c01 = c00, c10 = c00, c11 = c00;
    for (int ks = 0; ks < 8; ++ks) {
      v16bf bfr = load_b(wp_bf + (size_t)nc * 256 + ks * 32, 256, lane);
      c00 = wmma_bf(load_a(As[0] + ks * 32, 256, lane), bfr, c00);
      c01 = wmma_bf(load_a(As[1] + ks * 32, 256, lane), bfr, c01);
      c10 = wmma_bf(load_a(As[2] + ks * 32, 256, lane), bfr, c10);
      c11 = wmma_bf(load_a(As[3] + ks * 32, 256, lane), bfr, c11);
    }
    int cc = nc + (lane & 15);
    int pbase = pt * 16 + ((lane >> 4) << 3);
#pragma unroll
    for (int r = 0; r < 8; ++r) {
      int pg = pbase + r;
      float a0 = c00[r], a1 = c01[r], a2 = c10[r], a3 = c11[r];
      float av = 0.25f * (a0 + a1 + a2 + a3);
      float mm = fmaxf(fmaxf(a0, a1), fmaxf(a2, a3));
      avg[((size_t)b * 1024 + pg) * 512 + cc] = f2bf(av);
      mx [((size_t)b * 1024 + pg) * 512 + cc] = f2bf(mm);
      avgT[((size_t)b * 512 + cc) * 1024 + pg] = f2bf(av);
      mxT [((size_t)b * 512 + cc) * 1024 + pg] = f2bf(mm);
    }
  }
}

// ---------------- attention: one block = (b, 16-query tile); full softmax in LDS ----------------
// epilogue scatters the torch-faithful swapaxes/reshape: t[b, 2c+(m>>9), m&511] = o[b,m,c]
__global__ __launch_bounds__(256) void k_attn(const bf16* __restrict__ qkv,
                                              const bf16* __restrict__ kv,
                                              const bf16* __restrict__ kvT,
                                              bf16* __restrict__ tdst) {
  extern __shared__ char smem[];
  bf16*  Qs   = (bf16*)smem;                               // 16x512 bf16  = 16KB
  float* S    = (float*)(smem + 16384);                    // 16x1024 f32  = 64KB
  bf16*  P    = (bf16*)(smem + 16384 + 65536);             // 16x1024 bf16 = 32KB
  float* red  = (float*)(smem + 16384 + 65536 + 32768);    // 16x16 f32
  float* rowv = red + 256;                                 // 16 f32

  int blk = blockIdx.x;
  int b = blk >> 6, qt = blk & 63;
  int tid = threadIdx.x;
  const bf16* qsrc = qkv + ((size_t)b * 1024 + qt * 16) * 512;

  // stage Q tile into LDS (16B vector copies -> global_load_b128 + ds_store_b128)
  for (int it = tid; it < 1024; it += 256) {
    *(v8bf*)(Qs + it * 8) = *(const v8bf*)(qsrc + it * 8);
  }
  __syncthreads();

  int wave = tid >> 5, lane = tid & 31;
  // S = Q K^T * scale
  for (int kt = wave; kt < 64; kt += 8) {
    v8f acc = {0.f,0.f,0.f,0.f,0.f,0.f,0.f,0.f};
    const bf16* kbase = kv + ((size_t)b * 1024 + kt * 16) * 512;
    for (int ks = 0; ks < 16; ++ks) {
      v16bf a = load_a(Qs + ks * 32, 512, lane);
      v16bf bb = load_b(kbase + ks * 32, 512, lane);
      acc = wmma_bf(a, bb, acc);
    }
    int n = kt * 16 + (lane & 15);
    int mb = (lane >> 4) << 3;
#pragma unroll
    for (int r = 0; r < 8; ++r) S[(mb + r) * 1024 + n] = acc[r] * SCALE_ATTN;
  }
  __syncthreads();

  // row softmax (16 rows x 1024), P = bf16 probs
  {
    int row = tid >> 4, sub = tid & 15;
    float m = -3.4e38f;
    for (int j = sub; j < 1024; j += 16) m = fmaxf(m, S[row * 1024 + j]);
    red[row * 16 + sub] = m;
    __syncthreads();
    if (sub == 0) {
      float mm = red[row * 16];
      for (int t2 = 1; t2 < 16; ++t2) mm = fmaxf(mm, red[row * 16 + t2]);
      rowv[row] = mm;
    }
    __syncthreads();
    float mm = rowv[row];
    float s = 0.f;
    for (int j = sub; j < 1024; j += 16) {
      float e = __expf(S[row * 1024 + j] - mm);
      S[row * 1024 + j] = e;
      s += e;
    }
    red[row * 16 + sub] = s;
    __syncthreads();
    if (sub == 0) {
      float ss = 0.f;
      for (int t2 = 0; t2 < 16; ++t2) ss += red[row * 16 + t2];
      rowv[row] = 1.f / ss;
    }
    __syncthreads();
    float inv = rowv[row];
    for (int j = sub; j < 1024; j += 16) P[row * 1024 + j] = f2bf(S[row * 1024 + j] * inv);
  }
  __syncthreads();

  // O = P V  (V^T fragments are contiguous thanks to kvT), scatter-store permuted output
  for (int nt = wave; nt < 32; nt += 8) {
    v8f acc = {0.f,0.f,0.f,0.f,0.f,0.f,0.f,0.f};
    const bf16* vbase = kvT + ((size_t)b * 512 + nt * 16) * 1024;
    for (int ks = 0; ks < 32; ++ks) {
      v16bf a = load_a(P + ks * 32, 1024, lane);
      v16bf bb = load_b(vbase + ks * 32, 1024, lane);
      acc = wmma_bf(a, bb, acc);
    }
    int cc = nt * 16 + (lane & 15);
    int mb2 = qt * 16 + ((lane >> 4) << 3);
#pragma unroll
    for (int r = 0; r < 8; ++r) {
      int m = mb2 + r;
      tdst[((size_t)b * 1024 + 2 * cc + (m >> 9)) * 512 + (m & 511)] = f2bf(acc[r]);
    }
  }
}

// ---------------- fuse 1x1 conv GEMM: cat(t_self, beta*t_avg+(1-b)*t_max) @ fuse_w + bias ----------------
__global__ __launch_bounds__(256) void k_fuse(const bf16* __restrict__ tself,
                                              const bf16* __restrict__ tavg,
                                              const bf16* __restrict__ tmax,
                                              const bf16* __restrict__ fw_bf,
                                              const float* __restrict__ fuse_b,
                                              const float* __restrict__ beta,
                                              float* __restrict__ fusx,
                                              float* sum1, float* sqs1) {
  __shared__ bf16 As[16 * 1024];
  __shared__ float chS[512], chQ[512];
  int mt = blockIdx.x;
  int tid = threadIdx.x;
  for (int i = tid; i < 512; i += 256) { chS[i] = 0.f; chQ[i] = 0.f; }
  float bet = beta[0];
  for (int idx = tid; idx < 16 * 1024; idx += 256) {
    int row = idx >> 10, i = idx & 1023;
    size_t m = (size_t)mt * 16 + row;
    float v;
    if (i < 512) v = bf2f(tself[m * 512 + i]);
    else {
      float a = bf2f(tavg[m * 512 + (i - 512)]);
      float x2 = bf2f(tmax[m * 512 + (i - 512)]);
      v = bet * a + (1.f - bet) * x2;
    }
    As[idx] = f2bf(v);
  }
  __syncthreads();
  int wave = tid >> 5, lane = tid & 31;
  for (int t = 0; t < 4; ++t) {
    int nc = wave * 64 + t * 16;
    v8f acc = {0.f,0.f,0.f,0.f,0.f,0.f,0.f,0.f};
    for (int ks = 0; ks < 32; ++ks) {
      __builtin_prefetch(fw_bf + (size_t)nc * 1024 + (ks + 1) * 32, 0, 0);
      v16bf a = load_a(As + ks * 32, 1024, lane);
      v16bf bb = load_b(fw_bf + (size_t)nc * 1024 + ks * 32, 1024, lane);
      acc = wmma_bf(a, bb, acc);
    }
    int cc = nc + (lane & 15);
    int mb = mt * 16 + ((lane >> 4) << 3);
    float bia = fuse_b[cc];
    float ls = 0.f, lq = 0.f;
#pragma unroll
    for (int r = 0; r < 8; ++r) {
      float v = acc[r] + bia;
      fusx[(size_t)(mb + r) * 512 + cc] = v;
      ls += v;
      lq += v * v;
    }
    atomicAdd(&chS[cc], ls);
    atomicAdd(&chQ[cc], lq);
  }
  __syncthreads();
  for (int i = tid; i < 512; i += 256) {
    atomicAdd(&sum1[i], chS[i]);
    atomicAdd(&sqs1[i], chQ[i]);
  }
}

// ---------------- BN stats -> per-channel scale/shift ----------------
__global__ void k_stats(const float* sum, const float* sqs, const float* g, const float* bb,
                        float* A, float* B, float inv_cnt) {
  int c = threadIdx.x;
  float mean = sum[c] * inv_cnt;
  float var = sqs[c] * inv_cnt - mean * mean;
  float a = rsqrtf(var + 1e-5f) * g[c];
  A[c] = a;
  B[c] = bb[c] - mean * a;
}

// ---------------- x2 = x + gamma * relu(BN1(fusx))  -> bf16 for conv ----------------
__global__ void k_bn1(const float* __restrict__ x, const float* __restrict__ fusx,
                      const float* __restrict__ A, const float* __restrict__ B,
                      const float* __restrict__ gamma, bf16* __restrict__ x2) {
  size_t i = (size_t)blockIdx.x * blockDim.x + threadIdx.x;
  if (i >= (size_t)M_TOT * 512) return;
  int c = (int)(i & 511);
  float y = fmaxf(fusx[i] * A[c] + B[c], 0.f);
  x2[i] = f2bf(x[i] + gamma[0] * y);
}

// ---------------- 3x3 SAME conv as implicit GEMM (K = 9*512), bias + BN2 stats ----------------
__global__ __launch_bounds__(256) void k_conv(const bf16* __restrict__ x2,
                                              const bf16* __restrict__ ow_bf,
                                              const float* __restrict__ out_b,
                                              float* __restrict__ y,
                                              float* sum2, float* sqs2) {
  __shared__ float chS[512], chQ[512];
  int mt = blockIdx.x;
  int tid = threadIdx.x;
  for (int i = tid; i < 512; i += 256) { chS[i] = 0.f; chQ[i] = 0.f; }
  __syncthreads();
  int wave = tid >> 5, lane = tid & 31;
  int row = lane & 15;
  int m = mt * 16 + row;
  int b = m >> 10, h = (m >> 5) & 31, w = m & 31;
  int kb = (lane >> 4) << 3;
  for (int t = 0; t < 4; ++t) {
    int nc = wave * 64 + t * 16;
    v8f acc = {0.f,0.f,0.f,0.f,0.f,0.f,0.f,0.f};
    for (int ks = 0; ks < 144; ++ks) {
      int tap = ks >> 4;            // 16 K-steps per tap (512/32)
      int i0 = (ks & 15) << 5;
      int dy = tap / 3 - 1, dx = tap % 3 - 1;
      int hh = h + dy, ww = w + dx;
      v16bf a;
      if ((unsigned)hh < 32u && (unsigned)ww < 32u) {
        const bf16* ap = x2 + (((size_t)b * 32 + hh) * 32 + ww) * 512 + i0 + kb;
        v8bf lo = *(const v8bf*)ap;
        v8bf hi = *(const v8bf*)(ap + 16);
        a = __builtin_shufflevector(lo, hi, 0,1,2,3,4,5,6,7,8,9,10,11,12,13,14,15);
      } else {
        a = zero16();
      }
      __builtin_prefetch(ow_bf + ((size_t)tap * 512 + nc) * 512 + i0 + 512, 0, 0);
      v16bf bb = load_b(ow_bf + ((size_t)tap * 512 + nc) * 512 + i0, 512, lane);
      acc = wmma_bf(a, bb, acc);
    }
    int cc = nc + (lane & 15);
    int mb = mt * 16 + ((lane >> 4) << 3);
    float bia = out_b[cc];
    float ls = 0.f, lq = 0.f;
#pragma unroll
    for (int r = 0; r < 8; ++r) {
      float v = acc[r] + bia;
      y[(size_t)(mb + r) * 512 + cc] = v;
      ls += v;
      lq += v * v;
    }
    atomicAdd(&chS[cc], ls);
    atomicAdd(&chQ[cc], lq);
  }
  __syncthreads();
  for (int i = tid; i < 512; i += 256) {
    atomicAdd(&sum2[i], chS[i]);
    atomicAdd(&sqs2[i], chQ[i]);
  }
}

// ---------------- final BN2 + ReLU in place on d_out ----------------
__global__ void k_bn2(float* __restrict__ y, const float* __restrict__ A,
                      const float* __restrict__ B) {
  size_t i = (size_t)blockIdx.x * blockDim.x + threadIdx.x;
  if (i >= (size_t)M_TOT * 512) return;
  int c = (int)(i & 511);
  y[i] = fmaxf(y[i] * A[c] + B[c], 0.f);
}

// ---------------- host launcher ----------------
extern "C" void kernel_launch(void* const* d_in, const int* in_sizes, int n_in,
                              void* d_out, int out_size, void* d_ws, size_t ws_size,
                              hipStream_t stream) {
  (void)in_sizes; (void)n_in; (void)out_size; (void)ws_size;
  const float* x      = (const float*)d_in[0];
  const float* prevx  = (const float*)d_in[1];
  const float* w_prev = (const float*)d_in[2];
  const float* w_qkv  = (const float*)d_in[3];
  const float* fuse_w = (const float*)d_in[4];
  const float* fuse_b = (const float*)d_in[5];
  const float* bn1_g  = (const float*)d_in[6];
  const float* bn1_b  = (const float*)d_in[7];
  const float* out_w  = (const float*)d_in[8];
  const float* out_b  = (const float*)d_in[9];
  const float* bn2_g  = (const float*)d_in[10];
  const float* bn2_b  = (const float*)d_in[11];
  const float* gamma  = (const float*)d_in[12];
  const float* beta   = (const float*)d_in[13];
  float* out = (float*)d_out;

  char* ws = (char*)d_ws;
  size_t off = 0;
  auto alloc = [&](size_t bytes) -> void* {
    off = (off + 255) & ~(size_t)255;
    void* p = ws + off;
    off += bytes;
    return p;
  };

  const size_t Q = (size_t)M_TOT * 512;  // 8,388,608 elements
  bf16* qkv_bf = (bf16*)alloc(Q * 2);
  bf16* qkvT   = (bf16*)alloc(Q * 2);
  bf16* avg_bf = (bf16*)alloc(Q * 2);
  bf16* avgT   = (bf16*)alloc(Q * 2);
  bf16* max_bf = (bf16*)alloc(Q * 2);
  bf16* maxT   = (bf16*)alloc(Q * 2);
  bf16* t_self = (bf16*)alloc(Q * 2);
  bf16* t_avg  = (bf16*)alloc(Q * 2);
  bf16* t_max  = (bf16*)alloc(Q * 2);
  bf16* x2_bf  = (bf16*)alloc(Q * 2);
  float* fusx  = (float*)alloc(Q * 4);
  bf16* wq_bf  = (bf16*)alloc((size_t)512 * 512 * 2);
  bf16* wp_bf  = (bf16*)alloc((size_t)512 * 256 * 2);
  bf16* fw_bf  = (bf16*)alloc((size_t)512 * 1024 * 2);
  bf16* ow_bf  = (bf16*)alloc((size_t)9 * 512 * 512 * 2);
  float* stats = (float*)alloc((size_t)8 * 512 * 4);
  float* sum1 = stats,        *sqs1 = stats + 512;
  float* sum2 = stats + 1024, *sqs2 = stats + 1536;
  float* A1 = stats + 2048, *B1 = stats + 2560;
  float* A2 = stats + 3072, *B2 = stats + 3584;

  k_zero<<<8, 256, 0, stream>>>(stats, 2048);
  k_cvt<<<512, 256, 0, stream>>>(w_qkv, w_prev, fuse_w, out_w, wq_bf, wp_bf, fw_bf, ow_bf);
  k_gemm_qkv<<<1024, 256, 0, stream>>>(x, wq_bf, qkv_bf, qkvT);
  k_gemm_pool<<<1024, 256, 0, stream>>>(prevx, wp_bf, avg_bf, avgT, max_bf, maxT);

  const size_t attn_smem = 16384 + 65536 + 32768 + 1024 + 64;  // 115,776 B dynamic LDS
  k_attn<<<1024, 256, attn_smem, stream>>>(qkv_bf, qkv_bf, qkvT, t_self);
  k_attn<<<1024, 256, attn_smem, stream>>>(qkv_bf, avg_bf, avgT, t_avg);
  k_attn<<<1024, 256, attn_smem, stream>>>(qkv_bf, max_bf, maxT, t_max);

  k_fuse<<<1024, 256, 0, stream>>>(t_self, t_avg, t_max, fw_bf, fuse_b, beta, fusx, sum1, sqs1);
  k_stats<<<1, 512, 0, stream>>>(sum1, sqs1, bn1_g, bn1_b, A1, B1, 1.f / 16384.f);
  k_bn1<<<32768, 256, 0, stream>>>(x, fusx, A1, B1, gamma, x2_bf);

  k_conv<<<1024, 256, 0, stream>>>(x2_bf, ow_bf, out_b, out, sum2, sqs2);
  k_stats<<<1, 512, 0, stream>>>(sum2, sqs2, bn2_g, bn2_b, A2, B2, 1.f / 16384.f);
  k_bn2<<<32768, 256, 0, stream>>>(out, A2, B2);
}
